// EvolutionaryStructurePredictor_66984309948709
// MI455X (gfx1250) — matrix operand order repaired and checked by reading, not compiled, not run on hardware
//
#include <hip/hip_runtime.h>

// ---------------------------------------------------------------------------
// EvolutionaryStructurePredictor (Evoformer block) for gfx1250 (MI455X).
// Large GEMMs: v_wmma_f32_16x16x32_bf16, fp32->bf16 operands (v_perm_b32
// truncation pack), fp32 accum. Fast path is branch-free: weights are
// pre-transposed to [N,K] so both fragment loads are contiguous float4
// (global_load_b128); K padded to a multiple of 32 at repack time.
// Each wave computes a 32x32 register tile = 4 WMMAs per fragment-load group;
// the loop-invariant second-column-tile predicate is hoisted out of the K loop.
// LN / softmax / gating stay fp32 VALU.
// ---------------------------------------------------------------------------

#define NSEQ 128
#define LLEN 256
#define MSAD 256
#define PAIRD 128
#define COPM 12

typedef __bf16 bf16_t;
typedef __attribute__((ext_vector_type(16))) __bf16    v16bf;
typedef __attribute__((ext_vector_type(8)))  float     v8f;
typedef __attribute__((ext_vector_type(8)))  unsigned  v8u;

// Pack two fp32 into two bf16 (truncation) with a single v_perm_b32:
// dst = { lo[31:16], hi[31:16] }
__device__ inline unsigned pk2(float lo, float hi) {
  return __builtin_amdgcn_perm(__builtin_bit_cast(unsigned, hi),
                               __builtin_bit_cast(unsigned, lo),
                               0x07060302u);
}

__device__ inline v16bf frag_pack(float4 f0, float4 f1, float4 f2, float4 f3) {
  v8u u;
  u[0] = pk2(f0.x, f0.y); u[1] = pk2(f0.z, f0.w);
  u[2] = pk2(f1.x, f1.y); u[3] = pk2(f1.z, f1.w);
  u[4] = pk2(f2.x, f2.y); u[5] = pk2(f2.z, f2.w);
  u[6] = pk2(f3.x, f3.y); u[7] = pk2(f3.z, f3.w);
  return __builtin_bit_cast(v16bf, u);
}

// A fragment (16x32): lane r row = rowbase; elems 0..15 -> K half*16+0..15.
__device__ inline v16bf load_a_frag(const float* __restrict__ row, int k0, int half) {
  const float4* p0 = (const float4*)(row + k0 + half * 16);
  const float4* p1 = (const float4*)(row + k0 + half * 16 + 8);
  return frag_pack(p0[0], p0[1], p1[0], p1[1]);
}

// B fragment (32x16) from Bt[N,K]: lane r col = rowbase; elems -> K half*16+e.
__device__ inline v16bf load_b_frag(const float* __restrict__ row, int k0, int half) {
  const float4* p = (const float4*)(row + k0 + half * 16);
  return frag_pack(p[0], p[1], p[2], p[3]);
}

__device__ inline float act_apply(float v, int act) {
  if (act == 1) return fmaxf(v, 0.f);
  if (act == 2) return 1.f / (1.f + __expf(-v));
  return v;
}

// D tile store: lane r col fixed; acc elem i -> row m0 + half*8 + i.
__device__ inline void store_tile(v8f c, float* __restrict__ out, int ldo,
                                  int m0, int col, const float* __restrict__ bias,
                                  const float* __restrict__ residual,
                                  float alpha, int act, int half) {
  const float bv = bias ? bias[col] : 0.f;
#pragma unroll
  for (int i = 0; i < 8; ++i) {
    const size_t row = m0 + half * 8 + i;
    float v = act_apply(c[i] * alpha + bv, act);
    if (residual) v += residual[row * ldo + col];
    out[row * ldo + col] = v;
  }
}

// ---------------------------------------------------------------------------
// Fast WMMA GEMM: out[M,N] = act(alpha*A[M,K]@B + bias) + residual, B given
// TRANSPOSED as Bt[N,K]. Requirements: M%64==0, K%32==0, lda%4==0, N%16==0.
// blockDim=128 (4 waves, 2x2 over a 64x64 block tile); each wave computes a
// 32x32 register tile: 4 WMMAs sharing 2 A-frags and 2 B-frags per K chunk.
// ---------------------------------------------------------------------------
__global__ void wmma_gemm_fast(const float* __restrict__ A, int lda,
                               const float* __restrict__ Bt,
                               const float* __restrict__ bias,
                               const float* __restrict__ residual,
                               float* __restrict__ out, int ldo,
                               int M, int K, int N, float alpha, int act) {
  const int wave = threadIdx.x >> 5;
  const int lane = threadIdx.x & 31;
  const int half = lane >> 4;
  const int r    = lane & 15;
  const int m0 = blockIdx.x * 64 + (wave >> 1) * 32;
  const int n0 = blockIdx.y * 64 + (wave & 1) * 32;
  if (n0 >= N) return;                    // wave-uniform
  const bool two = (n0 + 16) < N;         // wave-uniform, loop-invariant

  const float* __restrict__ a0row = A  + (size_t)(m0 + r) * lda;
  const float* __restrict__ a1row = A  + (size_t)(m0 + 16 + r) * lda;
  const float* __restrict__ b0row = Bt + (size_t)(n0 + r) * K;
  const float* __restrict__ b1row = Bt + (size_t)(n0 + 16 + r) * K;

  v8f c00 = {}, c01 = {}, c10 = {}, c11 = {};
  if (two) {
    for (int k0 = 0; k0 < K; k0 += 32) {
      __builtin_prefetch(a0row + k0 + 32, 0, 0);
      __builtin_prefetch(b0row + k0 + 32, 0, 0);
      v16bf a0 = load_a_frag(a0row, k0, half);
      v16bf a1 = load_a_frag(a1row, k0, half);
      v16bf b0 = load_b_frag(b0row, k0, half);
      v16bf b1 = load_b_frag(b1row, k0, half);
      c00 = __builtin_amdgcn_wmma_f32_16x16x32_bf16(false, a0, false, b0,
                                                    (short)0, c00, false, false);
      c10 = __builtin_amdgcn_wmma_f32_16x16x32_bf16(false, a1, false, b0,
                                                    (short)0, c10, false, false);
      c01 = __builtin_amdgcn_wmma_f32_16x16x32_bf16(false, a0, false, b1,
                                                    (short)0, c01, false, false);
      c11 = __builtin_amdgcn_wmma_f32_16x16x32_bf16(false, a1, false, b1,
                                                    (short)0, c11, false, false);
    }
  } else {
    for (int k0 = 0; k0 < K; k0 += 32) {
      __builtin_prefetch(a0row + k0 + 32, 0, 0);
      __builtin_prefetch(b0row + k0 + 32, 0, 0);
      v16bf a0 = load_a_frag(a0row, k0, half);
      v16bf a1 = load_a_frag(a1row, k0, half);
      v16bf b0 = load_b_frag(b0row, k0, half);
      c00 = __builtin_amdgcn_wmma_f32_16x16x32_bf16(false, a0, false, b0,
                                                    (short)0, c00, false, false);
      c10 = __builtin_amdgcn_wmma_f32_16x16x32_bf16(false, a1, false, b0,
                                                    (short)0, c10, false, false);
    }
  }

  store_tile(c00, out, ldo, m0,      n0 + r,      bias, residual, alpha, act, half);
  store_tile(c10, out, ldo, m0 + 16, n0 + r,      bias, residual, alpha, act, half);
  if (two) {
    store_tile(c01, out, ldo, m0,      n0 + 16 + r, bias, residual, alpha, act, half);
    store_tile(c11, out, ldo, m0 + 16, n0 + 16 + r, bias, residual, alpha, act, half);
  }
}

// ---------------------------------------------------------------------------
// Small-N WMMA GEMM (N <= 16): B row-major [K,N], bounds-checked column loads.
// One wave per block; used only for the tiny bias/OPM projections.
// M%16==0, K%32==0 assumed.
// ---------------------------------------------------------------------------
__global__ void wmma_gemm_small(const float* __restrict__ A, int lda,
                                const float* __restrict__ W, int ldw,
                                const float* __restrict__ bias,
                                float* __restrict__ out, int ldo,
                                int M, int K, int N) {
  const int lane = threadIdx.x & 31;
  const int half = lane >> 4;
  const int r    = lane & 15;
  const int m0 = blockIdx.x * 16;
  const int n0 = blockIdx.y * 16;
  const int col = n0 + r;

  const float* __restrict__ arow = A + (size_t)(m0 + r) * lda;
  v8f c = {};
  for (int k0 = 0; k0 < K; k0 += 32) {
    v16bf a = load_a_frag(arow, k0, half);
    v8u bu;
#pragma unroll
    for (int e = 0; e < 8; ++e) {
      int k = k0 + half * 16 + 2 * e;
      float v0 = (col < N) ? W[(size_t)k * ldw + col] : 0.f;
      float v1 = (col < N) ? W[(size_t)(k + 1) * ldw + col] : 0.f;
      bu[e] = pk2(v0, v1);
    }
    v16bf b = __builtin_bit_cast(v16bf, bu);
    c = __builtin_amdgcn_wmma_f32_16x16x32_bf16(false, a, false, b,
                                                (short)0, c, false, false);
  }
  if (col < N) {
    const float bv = bias ? bias[col] : 0.f;
#pragma unroll
    for (int i = 0; i < 8; ++i) {
      const size_t row = m0 + half * 8 + i;
      out[row * ldo + col] = c[i] + bv;
    }
  }
}

// LayerNorm over last dim d; one row per block, blockDim = 256.
__global__ void ln_kernel(const float* __restrict__ x, const float* __restrict__ g,
                          const float* __restrict__ b, float* __restrict__ y, int d) {
  __shared__ float red[256];
  const size_t row = blockIdx.x;
  const float* xr = x + row * d;
  float s = 0.f;
  for (int i = threadIdx.x; i < d; i += 256) s += xr[i];
  red[threadIdx.x] = s; __syncthreads();
  for (int o = 128; o > 0; o >>= 1) {
    if (threadIdx.x < o) red[threadIdx.x] += red[threadIdx.x + o];
    __syncthreads();
  }
  float mu = red[0] / d; __syncthreads();
  float v = 0.f;
  for (int i = threadIdx.x; i < d; i += 256) { float t = xr[i] - mu; v += t * t; }
  red[threadIdx.x] = v; __syncthreads();
  for (int o = 128; o > 0; o >>= 1) {
    if (threadIdx.x < o) red[threadIdx.x] += red[threadIdx.x + o];
    __syncthreads();
  }
  float rstd = rsqrtf(red[0] / d + 1e-5f);
  for (int i = threadIdx.x; i < d; i += 256)
    y[row * d + i] = (xr[i] - mu) * rstd * g[i] + b[i];
}

// Fused gated attention with pairwise bias. Q/K/V/G layout [B, L, H*8];
// bias layout [L, L, H] indexed (query, key, h). One block per (b, h),
// blockDim = L = 256; K/V staged in LDS; online softmax.
__global__ void attn_kernel(const float* __restrict__ Q, const float* __restrict__ K,
                            const float* __restrict__ V, const float* __restrict__ G,
                            const float* __restrict__ bias, float* __restrict__ out,
                            int L, int H, float scale) {
  const int C = 8;
  __shared__ float Ks[256 * 8];
  __shared__ float Vs[256 * 8];
  const int b = blockIdx.x, h = blockIdx.y;
  const int HC = H * C;
  const size_t base = (size_t)b * L;
  for (int idx = threadIdx.x; idx < L * C; idx += 256) {
    int row = idx >> 3, c = idx & 7;
    Ks[idx] = K[(base + row) * HC + h * C + c];
    Vs[idx] = V[(base + row) * HC + h * C + c];
  }
  __syncthreads();

  const int q = threadIdx.x;
  float qv[8];
#pragma unroll
  for (int c = 0; c < 8; ++c) qv[c] = Q[(base + q) * HC + h * C + c];
  const float* brow = bias + (size_t)q * L * H + h;

  float mmax = -3.4e38f, lsum = 0.f;
  float acc[8] = {0.f, 0.f, 0.f, 0.f, 0.f, 0.f, 0.f, 0.f};
  for (int kk = 0; kk < L; ++kk) {
    float s = 0.f;
#pragma unroll
    for (int c = 0; c < 8; ++c) s += qv[c] * Ks[kk * 8 + c];
    s = s * scale + brow[(size_t)kk * H];
    float mnew = fmaxf(mmax, s);
    float corr = __expf(mmax - mnew);
    float p    = __expf(s - mnew);
    lsum = lsum * corr + p;
#pragma unroll
    for (int c = 0; c < 8; ++c) acc[c] = acc[c] * corr + p * Vs[kk * 8 + c];
    mmax = mnew;
  }
  float inv = 1.f / lsum;
#pragma unroll
  for (int c = 0; c < 8; ++c) {
    float gt = G[(base + q) * HC + h * C + c];
    out[(base + q) * HC + h * C + c] = acc[c] * inv * gt;
  }
}

// Weight transpose with K zero-padding: Wt[n*Kp + k] = (k<K) ? W[k*N+n] : 0
__global__ void transpose_pad_kernel(const float* __restrict__ W,
                                     float* __restrict__ Wt,
                                     int K, int N, int Kp) {
  int idx = blockIdx.x * 256 + threadIdx.x;
  if (idx >= N * Kp) return;
  int k = idx % Kp, n = idx / Kp;
  Wt[idx] = (k < K) ? W[(size_t)k * N + n] : 0.f;
}

// dst[(i*12+a)*128 + n] = src[(n*256 + i)*12 + a]  (OPM operand packing; used
// for both A ([3072,128]) and the transposed-B ([3072,128]) operands)
__global__ void opm_pack(const float* __restrict__ src, float* __restrict__ dst) {
  int idx = blockIdx.x * 256 + threadIdx.x;
  const int total = LLEN * COPM * NSEQ;        // 393216
  if (idx >= total) return;
  int n  = idx % NSEQ;
  int ia = idx / NSEQ;
  int i = ia / COPM, a = ia % COPM;
  dst[idx] = src[((size_t)n * LLEN + i) * COPM + a];
}

// ore[(i*256+j)*160 + ab] = ab<144 ? om[(i*12+a)*3072 + j*12+b] : 0   (K pad)
__global__ void opm_repack_outer(const float* __restrict__ om, float* __restrict__ ore) {
  int idx = blockIdx.x * 256 + threadIdx.x;
  const int total = LLEN * LLEN * 160;         // 10485760
  if (idx >= total) return;
  int ab = idx % 160;
  int ij = idx / 160;
  float v = 0.f;
  if (ab < COPM * COPM) {
    int j = ij % LLEN, i = ij / LLEN;
    int a = ab / COPM, b = ab % COPM;
    v = om[(size_t)(i * COPM + a) * (LLEN * COPM) + j * COPM + b];
  }
  ore[idx] = v;
}

// ---------------------------------------------------------------------------

static inline void gemm_fast(const float* A, int lda, const float* Bt,
                             const float* bias, const float* res,
                             float* out, int ldo,
                             int M, int K, int N, float alpha, int act,
                             hipStream_t s) {
  dim3 grid(M / 64, (N + 63) / 64);
  wmma_gemm_fast<<<grid, 128, 0, s>>>(A, lda, Bt, bias, res, out, ldo,
                                      M, K, N, alpha, act);
}

static inline void gemm_small(const float* A, int lda, const float* W, int ldw,
                              const float* bias, float* out, int ldo,
                              int M, int K, int N, hipStream_t s) {
  dim3 grid(M / 16, (N + 15) / 16);
  wmma_gemm_small<<<grid, 32, 0, s>>>(A, lda, W, ldw, bias, out, ldo, M, K, N);
}

static inline void tpose(const float* W, float* Wt, int K, int N, int Kp,
                         hipStream_t s) {
  int total = N * Kp;
  transpose_pad_kernel<<<(total + 255) / 256, 256, 0, s>>>(W, Wt, K, N, Kp);
}

extern "C" void kernel_launch(void* const* d_in, const int* in_sizes, int n_in,
                              void* d_out, int out_size, void* d_ws, size_t ws_size,
                              hipStream_t stream) {
  (void)in_sizes; (void)n_in; (void)out_size; (void)ws_size;
#define P(i) ((const float*)d_in[i])
  const float* msa_in  = P(0);
  const float* pair_in = P(1);

  const size_t NM = (size_t)NSEQ * LLEN * MSAD;   // 8388608
  float* out_msa  = (float*)d_out;
  float* out_pair = out_msa + NM;

  // Workspace (floats): WA 8.39M | WB 16.78M | WC 19.92M | WT 0.45M
  float* WA = (float*)d_ws;
  float* WB = WA + 8388608;
  float* WC = WB + 16777216;
  float* WT = WC + 19922944;

  // Transposed (and K-padded) weights
  float* wqT   = WT;            // [64,256]
  float* wkT   = WT + 16384;
  float* wvT   = WT + 32768;
  float* wgT   = WT + 49152;
  float* woT   = WT + 65536;    // [256,64]
  float* mt1T  = WT + 81920;    // [512,256]
  float* mt2T  = WT + 212992;   // [256,512]
  float* op3T  = WT + 344064;   // [128,160] (K=144 padded to 160)
  float* pqT   = WT + 364544;   // [32,128]
  float* pkT   = WT + 368640;
  float* pvT   = WT + 372736;
  float* pgT   = WT + 376832;
  float* pwoT  = WT + 380928;   // [128,32]
  float* pt1T  = WT + 385024;   // [256,128]
  float* pt2T  = WT + 417792;   // [128,256]

  tpose(P(4),  wqT,  256, 64,  256, stream);
  tpose(P(5),  wkT,  256, 64,  256, stream);
  tpose(P(6),  wvT,  256, 64,  256, stream);
  tpose(P(10), wgT,  256, 64,  256, stream);
  tpose(P(12), woT,  64,  256, 64,  stream);
  tpose(P(16), mt1T, 256, 512, 256, stream);
  tpose(P(18), mt2T, 512, 256, 512, stream);
  tpose(P(26), op3T, 144, 128, 160, stream);
  tpose(P(30), pqT,  128, 32,  128, stream);
  tpose(P(32), pkT,  128, 32,  128, stream);
  tpose(P(34), pvT,  128, 32,  128, stream);
  tpose(P(38), pgT,  128, 32,  128, stream);
  tpose(P(40), pwoT, 32,  128, 32,  stream);
  tpose(P(44), pt1T, 128, 256, 128, stream);
  tpose(P(46), pt2T, 256, 128, 256, stream);

  const float inv_sqrt8 = 0.35355339059327373f;

  // ---------------- Phase 1: MSA attention (pair-biased, gated) ------------
  ln_kernel<<<32768, 256, 0, stream>>>(msa_in, P(2), P(3), WA, MSAD);
  float* pz = WC + 4194304;
  ln_kernel<<<65536, 256, 0, stream>>>(pair_in, P(7), P(8), pz, PAIRD);

  float* q  = WB;
  float* k  = WB + 2097152;
  float* v  = WB + 4194304;
  float* gt = WB + 6291456;
  gemm_fast(WA, MSAD, wqT, nullptr, nullptr, q,  64, 32768, MSAD, 64, 1.f, 0, stream);
  gemm_fast(WA, MSAD, wkT, nullptr, nullptr, k,  64, 32768, MSAD, 64, 1.f, 0, stream);
  gemm_fast(WA, MSAD, wvT, nullptr, nullptr, v,  64, 32768, MSAD, 64, 1.f, 0, stream);
  gemm_fast(WA, MSAD, wgT, P(11),   nullptr, gt, 64, 32768, MSAD, 64, 1.f, 2, stream);

  float* bz = WC;                                   // [L, L, 8]
  gemm_small(pz, PAIRD, P(9), 8, nullptr, bz, 8, 65536, PAIRD, 8, stream);

  float* ao = WC + 1048576;                         // [128, 256, 64]
  attn_kernel<<<dim3(NSEQ, 8), 256, 0, stream>>>(q, k, v, gt, bz, ao, LLEN, 8, inv_sqrt8);
  gemm_fast(ao, 64, woT, P(13), msa_in, out_msa, MSAD, 32768, 64, MSAD, 1.f, 0, stream);

  // ---------------- Phase 2: MSA transition --------------------------------
  ln_kernel<<<32768, 256, 0, stream>>>(out_msa, P(14), P(15), WA, MSAD);
  gemm_fast(WA, MSAD, mt1T, P(17), nullptr, WB, 512, 32768, MSAD, 512, 1.f, 1, stream);
  gemm_fast(WB, 512, mt2T, P(19), out_msa, out_msa, MSAD, 32768, 512, MSAD, 1.f, 0, stream);

  // ---------------- Phase 3: outer product mean ----------------------------
  ln_kernel<<<32768, 256, 0, stream>>>(out_msa, P(20), P(21), WA, MSAD);
  float* a_nat = WB;                 // [128*256, 12]  proj_a = linear2
  float* b_nat = WB + 524288;        // [128*256, 12]  proj_b = linear1
  float* at    = WB + 1048576;       // [3072, 128]
  float* bt    = WB + 1572864;       // [3072, 128]  (transposed B operand)
  gemm_small(WA, MSAD, P(24), COPM, P(25), a_nat, COPM, 32768, MSAD, COPM, stream);
  gemm_small(WA, MSAD, P(22), COPM, P(23), b_nat, COPM, 32768, MSAD, COPM, stream);
  opm_pack<<<(393216 + 255) / 256, 256, 0, stream>>>(a_nat, at);
  opm_pack<<<(393216 + 255) / 256, 256, 0, stream>>>(b_nat, bt);

  float* om  = WC;                   // [3072, 3072] = at @ bt^T / 128
  float* ore = WC + 9437184;         // [65536, 160] (144 + 16 zero pad)
  gemm_fast(at, NSEQ, bt, nullptr, nullptr, om, LLEN * COPM,
            LLEN * COPM, NSEQ, LLEN * COPM, 1.f / (float)NSEQ, 0, stream);
  opm_repack_outer<<<(10485760 + 255) / 256, 256, 0, stream>>>(om, ore);
  gemm_fast(ore, 160, op3T, P(27), pair_in, out_pair, PAIRD,
            65536, 160, PAIRD, 1.f, 0, stream);

  // ---------------- Phase 4: pair attention --------------------------------
  ln_kernel<<<65536, 256, 0, stream>>>(out_pair, P(28), P(29), WA, PAIRD);
  float* pq = WB;
  float* pk = WB + 2097152;
  float* pv = WB + 4194304;
  float* pg = WB + 6291456;
  gemm_fast(WA, PAIRD, pqT, P(31), nullptr, pq, 32, 65536, PAIRD, 32, 1.f, 0, stream);
  gemm_fast(WA, PAIRD, pkT, P(33), nullptr, pk, 32, 65536, PAIRD, 32, 1.f, 0, stream);
  gemm_fast(WA, PAIRD, pvT, P(35), nullptr, pv, 32, 65536, PAIRD, 32, 1.f, 0, stream);
  gemm_fast(WA, PAIRD, pgT, P(39), nullptr, pg, 32, 65536, PAIRD, 32, 1.f, 2, stream);

  float* pb = WC;                                   // [L, L, 4]
  gemm_small(WA, PAIRD, P(36), 4, P(37), pb, 4, 65536, PAIRD, 4, stream);

  float* pao = WC + 1048576;                        // [256, 256, 32]
  attn_kernel<<<dim3(LLEN, 4), 256, 0, stream>>>(pq, pk, pv, pg, pb, pao, LLEN, 4, inv_sqrt8);
  gemm_fast(pao, 32, pwoT, P(41), out_pair, out_pair, PAIRD, 65536, 32, PAIRD, 1.f, 0, stream);

  // ---------------- Phase 5: pair transition -------------------------------
  ln_kernel<<<65536, 256, 0, stream>>>(out_pair, P(42), P(43), WA, PAIRD);
  gemm_fast(WA, PAIRD, pt1T, P(45), nullptr, WB, 256, 65536, PAIRD, 256, 1.f, 1, stream);
  gemm_fast(WB, 256, pt2T, P(47), out_pair, out_pair, PAIRD, 65536, 256, PAIRD, 1.f, 0, stream);
#undef P
}
